// HAGSM_57655640981781
// MI455X (gfx1250) — matrix operand order repaired and checked by reading, not compiled, not run on hardware
//
#include <hip/hip_runtime.h>
#include <hip/hip_bf16.h>
#include <math.h>

typedef _Float16 f16_t;
typedef __attribute__((ext_vector_type(16))) _Float16 v16h;
typedef __attribute__((ext_vector_type(8)))  _Float16 v8h;
typedef __attribute__((ext_vector_type(8)))  float    v8f;

#define C_CH 256
#define Hh   56
#define Ww   56
#define Tt   8
#define Bb   4
#define BTOT 32
#define HW   3136
#define CB   64     // channel staging block
#define XPAD 72     // padded channel row (bank decorrelation)

// ---------------------------------------------------------------- helpers
__device__ __forceinline__ v16h make_a(const f16_t* arow, int cbl, int hk) {
  // A 16x32 f16, contiguous k order used here (B loads match): lane<16 -> K {0..15}; lane>=16 -> K {16..31}
  v8h alo = *(const v8h*)(arow + cbl + hk * 16);
  v8h ahi = *(const v8h*)(arow + cbl + hk * 16 + 8);
  v16h av;
#pragma unroll
  for (int i = 0; i < 8; ++i) { av[i] = alo[i]; av[8 + i] = ahi[i]; }
  return av;
}

// ---------------------------------------------------------------- kernel 0: zero BN sums
__global__ void k_init(float* __restrict__ sums) {
  for (int i = threadIdx.x; i < 512; i += 256) sums[i] = 0.0f;
}

// ---------------------------------------------------------------- kernel 1: attn conv (WMMA) + scale + BN partial sums
__global__ __launch_bounds__(128) void k_attn(const float* __restrict__ x,
                                              const float* __restrict__ attn_w,
                                              const float* __restrict__ attn_b,
                                              float* __restrict__ xh,
                                              float* __restrict__ sums) {
  __shared__ __align__(32) f16_t lds_x[3 * 58 * XPAD];   // 25.0 KB
  __shared__ __align__(32) f16_t lds_w[9 * 5 * 264];     // 23.8 KB (slot 4 = zeros)
  __shared__ float lds_attn[64 * 4];

  const int h    = blockIdx.x;
  const int bt   = blockIdx.y;
  const int tid  = threadIdx.x;
  const int lane = tid & 31;
  const int wave = tid >> 5;

  // Stage attn weights as wB[tap][oc_slot][ci] (K = ci contiguous), slot 4 = zeros for N-pad lanes.
  for (int idx = tid; idx < 9 * 4 * 256; idx += 128) {
    const int tap = idx >> 10;
    const int oc  = (idx >> 8) & 3;
    const int ci  = idx & 255;
    const int kh = tap / 3, kw = tap % 3;
    lds_w[(tap * 5 + oc) * 264 + ci] = (f16_t)attn_w[((oc * 256 + ci) * 3 + kh) * 3 + kw];
  }
  for (int idx = tid; idx < 9 * 256; idx += 128)
    lds_w[((idx >> 8) * 5 + 4) * 264 + (idx & 255)] = (f16_t)0.0f;

  v8f acc = {};
  const int w0 = wave * 16;           // one 16-pixel tile per wave (tiles at w0=0,16,32,48)
  const int m  = lane & 15;
  const int hk = lane >> 4;

  for (int cb0 = 0; cb0 < 256; cb0 += CB) {
    __syncthreads();
    {  // stage x[ci-block][h-1..h+1][w-1..w+56] as f16, zero-padded
      const int ciq = tid >> 6;       // 0..1
      const int cq  = tid & 63;
      if (cq < 58) {
        const int w = cq - 1;
        for (int r = 0; r < 3; ++r) {
          const int hh = h - 1 + r;
          for (int c2 = 0; c2 < CB; c2 += 2) {
            const int ci = cb0 + c2 + ciq;
            float v = 0.0f;
            if (hh >= 0 && hh < Hh && w >= 0 && w < Ww)
              v = x[((bt * C_CH + ci) * Hh + hh) * Ww + w];
            lds_x[(r * 58 + cq) * XPAD + c2 + ciq] = (f16_t)v;
          }
        }
      }
    }
    __syncthreads();
    for (int tap = 0; tap < 9; ++tap) {
      const int kh = tap / 3, kw = tap % 3;
      int col = w0 + m + kw; if (col > 57) col = 57;    // pix>=56 are discarded later
      const f16_t* arow = &lds_x[(kh * 58 + col) * XPAD];
      const int slot    = (m < 4) ? m : 4;
      const f16_t* brow = &lds_w[(tap * 5 + slot) * 264 + cb0];
      for (int cbl = 0; cbl < CB; cbl += 32) {
        v16h av = make_a(arow, cbl, hk);
        v16h bv = *(const v16h*)(brow + cbl + hk * 16);
        acc = __builtin_amdgcn_wmma_f32_16x16x32_f16(false, av, false, bv,
                                                     (short)0, acc, false, false);
      }
    }
  }

  // D layout: vgpr r -> pixel r + 8*(lane>=16); lane&15 -> oc
  if (m < 4) {
    const float bsc = attn_b[m];
#pragma unroll
    for (int r = 0; r < 8; ++r) {
      const int pix = w0 + r + hk * 8;
      float v = fmaxf(acc[r] + bsc, 0.0f);
      lds_attn[pix * 4 + m] = 1.0f / (1.0f + __expf(-v));
    }
  }
  __syncthreads();

  // xh = x * attn[head]; accumulate per-channel sum / sumsq (wave reduce + atomics)
  for (int ci = wave; ci < C_CH; ci += 4) {
    const int head = ci >> 6;
    float s = 0.0f, s2 = 0.0f;
    for (int w = lane; w < Ww; w += 32) {
      const float v = x[((bt * C_CH + ci) * Hh + h) * Ww + w] * lds_attn[w * 4 + head];
      xh[((bt * C_CH + ci) * Hh + h) * Ww + w] = v;
      s += v; s2 += v * v;
    }
#pragma unroll
    for (int off = 16; off > 0; off >>= 1) {
      s  += __shfl_down(s, off, 32);
      s2 += __shfl_down(s2, off, 32);
    }
    if (lane == 0) {
      atomicAdd(&sums[ci], s);
      atomicAdd(&sums[256 + ci], s2);
    }
  }
}

// ---------------------------------------------------------------- kernel 2: BN finalize -> scale/bias
__global__ void k_bn(const float* __restrict__ sums, const float* __restrict__ gamma,
                     const float* __restrict__ beta, float* __restrict__ scb) {
  const int c = threadIdx.x;
  if (c < 256) {
    const float N  = (float)(Bb * Tt * HW);
    const float mu = sums[c] / N;
    const float var = sums[256 + c] / N - mu * mu;
    const float sc  = gamma[c] * rsqrtf(var + 1e-5f);
    scb[c]       = sc;
    scb[256 + c] = beta[c] - mu * sc;
  }
}

// ---------------------------------------------------------------- kernel 3: packed conv3d gates (WMMA) + tanh
// N packs 16 combos: n = f*8 + g*4 + (tau&3); two tau-groups handled by wave pairs.
// Reduction K iterates (t_abs, kh, kw, ci); lane B fragment is zero unless
// dt = t_abs - tau is in {-d, 0, d} and the ci block is in this gate's channel half.
__global__ __launch_bounds__(256) void k_gates(const float* __restrict__ xh,
                                               const float* __restrict__ scb,
                                               const float* __restrict__ gate_w0,
                                               const float* __restrict__ gate_b0,
                                               const float* __restrict__ gate_w1,
                                               const float* __restrict__ gate_b1,
                                               float* __restrict__ gates) {
  __shared__ __align__(32) f16_t lds_x[3 * 58 * XPAD];   // 25.0 KB
  __shared__ __align__(32) f16_t lds_w[108 * 136];       // 29.4 KB: [fg][kt][khw][ci(128)]
  __shared__ __align__(32) f16_t zblk[32];

  const int h    = blockIdx.x;
  const int b    = blockIdx.y;
  const int tid  = threadIdx.x;
  const int lane = tid & 31;
  const int wave = tid >> 5;
  const int w0      = (wave >> 1) * 16;
  const int tgroup  = wave & 1;

  const int n   = lane & 15;
  const int hk  = lane >> 4;
  const int f   = n >> 3;
  const int g   = (n >> 2) & 1;
  const int tau = tgroup * 4 + (n & 3);
  const int d   = f + 1;

  // Stage both gate weight tensors, K = ci contiguous per (f,g,kt,kh,kw)
  for (int idx = tid; idx < 2 * 2 * 128 * 27; idx += 256) {
    const int fw = idx / (2 * 128 * 27);
    int r = idx - fw * (2 * 128 * 27);
    const int gw = r / (128 * 27);
    r -= gw * (128 * 27);
    const int ci  = r / 27;
    const int tap = r - ci * 27;          // kt*9 + khw
    const int kt  = tap / 9;
    const int khw = tap - kt * 9;
    const float* wp = fw ? gate_w1 : gate_w0;
    lds_w[(((fw * 2 + gw) * 3 + kt) * 9 + khw) * 136 + ci] =
        (f16_t)wp[(gw * 128 + ci) * 27 + tap];
  }
  if (tid < 32) zblk[tid] = (f16_t)0.0f;

  v8f acc = {};
  const int m = n;   // pixel row in A

  for (int ta = 0; ta < Tt; ++ta) {
    const int dt = ta - tau;
    int kt = -1;
    if (dt == 0) kt = 1; else if (dt == d) kt = 2; else if (dt == -d) kt = 0;
    const int ktc = (kt < 0) ? 0 : kt;

    for (int cb0 = 0; cb0 < 256; cb0 += CB) {
      __syncthreads();
      {  // stage xr = relu(xh*scale+bias) slice for time ta, channel block
        const int ciq = tid >> 6;   // 0..3
        const int cq  = tid & 63;
        if (cq < 58) {
          const int w = cq - 1;
          for (int r = 0; r < 3; ++r) {
            const int hh = h - 1 + r;
            for (int c4 = 0; c4 < CB; c4 += 4) {
              const int ci = cb0 + c4 + ciq;
              float v = 0.0f;
              if (hh >= 0 && hh < Hh && w >= 0 && w < Ww) {
                const float xv = xh[(((b * Tt + ta) * C_CH + ci) * Hh + hh) * Ww + w];
                v = fmaxf(xv * scb[ci] + scb[256 + ci], 0.0f);
              }
              lds_x[(r * 58 + cq) * XPAD + c4 + ciq] = (f16_t)v;
            }
          }
        }
      }
      __syncthreads();

      const bool gok = (kt >= 0) && ((cb0 >> 7) == g);
      for (int tap = 0; tap < 9; ++tap) {
        const int kh = tap / 3, kw = tap % 3;
        int col = w0 + m + kw; if (col > 57) col = 57;
        const f16_t* arow = &lds_x[(kh * 58 + col) * XPAD];
        const f16_t* brow =
            &lds_w[(((f * 2 + g) * 3 + ktc) * 9 + tap) * 136 + (cb0 & 127) + hk * 16];
        for (int cbl = 0; cbl < CB; cbl += 32) {
          v16h av = make_a(arow, cbl, hk);
          const f16_t* bp = gok ? (brow + cbl) : zblk;
          v16h bv = *(const v16h*)bp;
          acc = __builtin_amdgcn_wmma_f32_16x16x32_f16(false, av, false, bv,
                                                       (short)0, acc, false, false);
        }
      }
    }
  }

  const float gb = (f ? gate_b1 : gate_b0)[g];
#pragma unroll
  for (int r = 0; r < 8; ++r) {
    const int w = w0 + r + hk * 8;
    if (w < Ww) {
      const float gv = tanhf(acc[r] + gb);
      gates[(((f * 2 + g) * Bb + b) * Tt + tau) * HW + h * Ww + w] = gv;
    }
  }
}

// ---------------------------------------------------------------- kernel 4: temporal shift + shuffle fuse -> out
__global__ __launch_bounds__(256) void k_fuse(const float* __restrict__ xh,
                                              const float* __restrict__ scb,
                                              const float* __restrict__ gates,
                                              const float* __restrict__ alpha,
                                              float* __restrict__ out) {
  const int bt = blockIdx.x >> 8;
  const int c  = blockIdx.x & 255;
  const int b = bt >> 3, t = bt & 7;
  const int half = c >> 7;                       // 0: y1 path (shift-left), 1: y2 (shift-right)
  const int cc   = c & 127;
  const int src  = half * 128 + (cc & 1) * 64 + (cc >> 1);   // inverse channel shuffle
  const float sc = scb[src], bi = scb[256 + src];
  const float a0 = alpha[0], a1 = alpha[1];
  const float* xrow = xh + ((size_t)(bt * C_CH) + src) * HW;
  float* orow       = out + ((size_t)(bt * C_CH) + c) * HW;

  for (int p = threadIdx.x; p < HW; p += 256) {
    __builtin_prefetch(xrow + p + 256, 0, 1);
    const float x0 = fmaxf(xrow[p] * sc + bi, 0.0f);
    float o = 0.0f;
#pragma unroll
    for (int f = 0; f < 2; ++f) {
      const int d  = f + 1;
      const int fg = f * 2 + half;
      const float gt = gates[((fg * Bb + b) * Tt + t) * HW + p];
      float y = x0 - gt * x0;
      const int tt = half ? (t - d) : (t + d);
      if (tt >= 0 && tt < Tt) {
        const float xs =
            fmaxf(xh[((size_t)((b * Tt + tt) * C_CH) + src) * HW + p] * sc + bi, 0.0f);
        const float gs = gates[((fg * Bb + b) * Tt + tt) * HW + p];
        y += gs * xs;
      }
      o += (f ? a1 : a0) * y;
    }
    orow[p] = o;
  }
}

// ---------------------------------------------------------------- launch
extern "C" void kernel_launch(void* const* d_in, const int* in_sizes, int n_in,
                              void* d_out, int out_size, void* d_ws, size_t ws_size,
                              hipStream_t stream) {
  (void)in_sizes; (void)n_in; (void)out_size; (void)ws_size;
  const float* x      = (const float*)d_in[0];
  const float* attn_w = (const float*)d_in[1];
  const float* attn_b = (const float*)d_in[2];
  const float* bng    = (const float*)d_in[3];
  const float* bnb    = (const float*)d_in[4];
  const float* gw0    = (const float*)d_in[5];
  const float* gb0    = (const float*)d_in[6];
  const float* gw1    = (const float*)d_in[7];
  const float* gb1    = (const float*)d_in[8];
  const float* alpha  = (const float*)d_in[9];
  float* out = (float*)d_out;

  float* xh    = (float*)d_ws;                         // 32*256*3136 f32
  float* sums  = xh + (size_t)BTOT * C_CH * HW;        // sum[256], sumsq[256]
  float* scb   = sums + 512;                           // scale[256], bias[256]
  float* gates = scb + 512;                            // 4 * 4 * 8 * 3136 f32

  k_init <<<dim3(1),          dim3(256), 0, stream>>>(sums);
  k_attn <<<dim3(Hh, BTOT),   dim3(128), 0, stream>>>(x, attn_w, attn_b, xh, sums);
  k_bn   <<<dim3(1),          dim3(256), 0, stream>>>(sums, bng, bnb, scb);
  k_gates<<<dim3(Hh, Bb),     dim3(256), 0, stream>>>(xh, scb, gw0, gb0, gw1, gb1, gates);
  k_fuse <<<dim3(BTOT * C_CH), dim3(256), 0, stream>>>(xh, scb, gates, alpha, out);
}
